// HOIModel_32195074851418
// MI455X (gfx1250) — matrix operand order repaired and checked by reading, not compiled, not run on hardware
//
#include <hip/hip_runtime.h>

// ---------------------------------------------------------------------------
// HOI model for MI455X (gfx1250, wave32):
//   1) ROI mean pooling (bandwidth-bound, VALU)
//   2) per-group argsort of scores (tiny)
//   3) pair construction -> bf16 activation matrix X [3072, 1536]
//   4) 3x GEMM via V_WMMA_F32_16X16X32_BF16 with LDS tiling staged by
//      GLOBAL_LOAD_ASYNC_TO_LDS_B32 (ASYNCcnt), fused bias/ReLU epilogue
// ---------------------------------------------------------------------------

typedef __attribute__((ext_vector_type(16))) __bf16 v16bf;
typedef __attribute__((ext_vector_type(8)))  float  v8f;

union Frag16 { v16bf v; unsigned int u[8]; };

__device__ __forceinline__ unsigned short f2bf(float f) {
    unsigned int u = __float_as_uint(f);
    u += 0x7FFFu + ((u >> 16) & 1u);      // round-to-nearest-even
    return (unsigned short)(u >> 16);
}

// Async memory->LDS copy (no VGPR staging). VDST = logical LDS byte offset
// (low 32 bits of the flat pointer into the LDS aperture), VADDR = 64-bit
// global address. Tracked by ASYNCcnt.
__device__ __forceinline__ void async_copy_b32(unsigned lds_off,
                                               const unsigned int* gsrc) {
    asm volatile("global_load_async_to_lds_b32 %0, %1, off"
                 :
                 : "v"(lds_off), "v"(gsrc)
                 : "memory");
}
__device__ __forceinline__ void wait_asynccnt0() {
    asm volatile("s_wait_asynccnt 0x0" ::: "memory");
}

#define B_    16
#define D_    32
#define C_    768
#define NH_   8
#define NO_   24
#define PAIRS 192                 // NH_*NO_
#define MROWS 3072                // B_*PAIRS

// ---------------------------------------------------------------------------
// 1) ROI mean pooling: one block per (b,d); 256 threads cover 768 channels x3
// ---------------------------------------------------------------------------
__global__ __launch_bounds__(256)
void roi_pool_kernel(const float* __restrict__ feat,
                     const float* __restrict__ boxes,
                     float* __restrict__ pooled) {
    const int bd = blockIdx.x;                 // b*32 + d
    const int b  = bd >> 5;
    const float4 box = ((const float4*)boxes)[bd];
    const float s = 64.0f;                     // IMG/PATCH = 896/14
    int x1 = (int)floorf((box.x - box.z * 0.5f) * s);
    int y1 = (int)floorf((box.y - box.w * 0.5f) * s);
    int x2 = (int)floorf((box.x + box.z * 0.5f) * s);
    int y2 = (int)floorf((box.y + box.w * 0.5f) * s);
    x1 = max(0, min(64, x1)); x2 = max(0, min(64, x2));
    y1 = max(0, min(64, y1)); y2 = max(0, min(64, y2));
    const int area = (y2 - y1) * (x2 - x1);
    const float inv = 1.0f / (float)max(1, area);
    const size_t fb = (size_t)b * 64 * 64 * C_;
    const int c = threadIdx.x;                 // channels c, c+256, c+512
    float a0 = 0.f, a1 = 0.f, a2 = 0.f;
    for (int y = y1; y < y2; ++y) {
        const float* rowp = feat + fb + ((size_t)y * 64) * C_ + c;
        for (int x = x1; x < x2; ++x) {
            const float* p = rowp + (size_t)x * C_;
            a0 += p[0]; a1 += p[256]; a2 += p[512];
        }
    }
    float* o = pooled + (size_t)bd * C_ + c;
    o[0] = a0 * inv; o[256] = a1 * inv; o[512] = a2 * inv;
}

// ---------------------------------------------------------------------------
// 2) descending argsort (selection sort, not stable) of 8 human / 24 object scores per batch
// ---------------------------------------------------------------------------
__device__ void argsort_desc(const float* __restrict__ s, int n, int* __restrict__ out) {
    float v[NO_]; int id[NO_];
    for (int i = 0; i < n; ++i) { v[i] = s[i]; id[i] = i; }
    for (int i = 0; i < n; ++i) {
        int best = i;
        for (int j = i + 1; j < n; ++j) if (v[j] > v[best]) best = j;
        float tv = v[i]; v[i] = v[best]; v[best] = tv;
        int   ti = id[i]; id[i] = id[best]; id[best] = ti;
    }
    for (int i = 0; i < n; ++i) out[i] = id[i];
}

__global__ void argsort_kernel(const float* __restrict__ scores,
                               int* __restrict__ hidx, int* __restrict__ oidx) {
    if (threadIdx.x != 0) return;
    const int b = blockIdx.x;
    argsort_desc(scores + b * D_,       NH_, hidx + b * NH_);
    argsort_desc(scores + b * D_ + NH_, NO_, oidx + b * NO_);
}

// ---------------------------------------------------------------------------
// 3a) weight convert + transpose:  w[K][N] f32  ->  wt[Npad][K] bf16
// ---------------------------------------------------------------------------
__global__ void convert_wT_kernel(const float* __restrict__ w,
                                  unsigned short* __restrict__ wt,
                                  int K, int N, int Npad) {
    const int idx = blockIdx.x * 256 + threadIdx.x;
    if (idx >= Npad * K) return;
    const int n = idx / K, k = idx % K;
    const float v = (n < N) ? w[(size_t)k * N + n] : 0.0f;
    wt[idx] = f2bf(v);
}

// ---------------------------------------------------------------------------
// 3b) pair gather:  X[b, i*24+j, 0:768]=hfeat, [768:1536]=ofeat  (bf16)
// ---------------------------------------------------------------------------
__global__ __launch_bounds__(256)
void build_pairs_kernel(const float* __restrict__ pooled,
                        const int* __restrict__ hidx, const int* __restrict__ oidx,
                        unsigned short* __restrict__ X) {
    const int row = blockIdx.x;                // b*192 + i*24 + j
    const int b = row / PAIRS, r = row % PAIRS;
    const int i = r / NO_, j = r % NO_;
    const int hd = hidx[b * NH_ + i];
    const int od = NH_ + oidx[b * NO_ + j];
    const float* ph = pooled + ((size_t)b * D_ + hd) * C_;
    const float* po = pooled + ((size_t)b * D_ + od) * C_;
    unsigned short* xr = X + (size_t)row * (2 * C_);
    for (int c = threadIdx.x; c < C_; c += 256) {
        xr[c]        = f2bf(ph[c]);
        xr[C_ + c]   = f2bf(po[c]);
    }
}

// ---------------------------------------------------------------------------
// 4) bf16 WMMA GEMM:  C[M][ldc] = act(A[M][K] * BT[N][K]^T + bias)
//    block tile 128x64, 8 waves (4x2), wave tile 32x32 = 2x2 WMMA, K-tile 32
//    Tiles staged via GLOBAL_LOAD_ASYNC_TO_LDS_B32 (no VGPR staging).
// ---------------------------------------------------------------------------
#define TM 128
#define TN 64
#define TK 32
#define LDSA_OFF 0
#define LDSB_OFF (TM * 17)

template <bool RELU, bool OUT_BF16>
__global__ __launch_bounds__(256)
void gemm_bf16_kernel(const unsigned short* __restrict__ A,   // [M][K] bf16
                      const unsigned short* __restrict__ BT,  // [Npad][K] bf16
                      const float* __restrict__ bias,         // [>=Nstore] f32
                      void* __restrict__ Cout,                // [M][ldc]
                      int K, int Nstore, int ldc) {
    __shared__ unsigned int lds[TM * 17 + TN * 17];  // rows padded to 17 dwords

    const int mBase = blockIdx.x * TM;
    const int nBase = blockIdx.y * TN;
    const int Ku = K >> 1;                    // dwords per bf16 row

    const unsigned int* Au = (const unsigned int*)A;
    const unsigned int* Bu = (const unsigned int*)BT;

    const int wave  = threadIdx.x >> 5;
    const int waveM = wave >> 1;              // 0..3
    const int waveN = wave & 1;               // 0..1
    const int lane  = threadIdx.x & 31;
    const int hi    = lane >> 4;              // lane half
    const int lm    = lane & 15;

    const v8f zero = {0.f, 0.f, 0.f, 0.f, 0.f, 0.f, 0.f, 0.f};
    v8f acc[2][2] = {{zero, zero}, {zero, zero}};

    for (int k0 = 0; k0 < K; k0 += TK) {
        __syncthreads();                      // prior frag reads done (wmma consumed them)
        const int ku0 = k0 >> 1;
        // stage A tile: 128 rows x 16 dwords, async -> LDS
#pragma unroll
        for (int i = 0; i < 8; ++i) {
            const int idx = threadIdx.x + 256 * i;
            const int r = idx >> 4, c = idx & 15;
            async_copy_b32((unsigned)(size_t)&lds[LDSA_OFF + r * 17 + c],
                           Au + (size_t)(mBase + r) * Ku + ku0 + c);
        }
        // stage B tile (N-major): 64 rows x 16 dwords, async -> LDS
#pragma unroll
        for (int i = 0; i < 4; ++i) {
            const int idx = threadIdx.x + 256 * i;
            const int r = idx >> 4, c = idx & 15;
            async_copy_b32((unsigned)(size_t)&lds[LDSB_OFF + r * 17 + c],
                           Bu + (size_t)(nBase + r) * Ku + ku0 + c);
        }
        wait_asynccnt0();
        __syncthreads();

        // A fragments: per CDNA5 ISA 16-bit A 16x32 layout
        Frag16 af[2];
#pragma unroll
        for (int mi = 0; mi < 2; ++mi) {
            const unsigned int* base = &lds[LDSA_OFF + (waveM * 32 + mi * 16 + lm) * 17];
#pragma unroll
            for (int p = 0; p < 8; ++p) {
                const int kp = p + 4 * hi + ((p >= 4) ? 4 : 0);
                af[mi].u[p] = base[kp];
            }
        }
        // B fragments: N striped over lanes; dword v holds K=2v+8*hi (+8 for v>=4)
        Frag16 bfr[2];
#pragma unroll
        for (int ni = 0; ni < 2; ++ni) {
            const unsigned int* base = &lds[LDSB_OFF + (waveN * 32 + ni * 16 + lm) * 17];
#pragma unroll
            for (int v = 0; v < 8; ++v)
                bfr[ni].u[v] = base[v + 4 * hi + ((v >= 4) ? 4 : 0)];
        }
#pragma unroll
        for (int mi = 0; mi < 2; ++mi)
#pragma unroll
            for (int ni = 0; ni < 2; ++ni)
                acc[mi][ni] = __builtin_amdgcn_wmma_f32_16x16x32_bf16(
                    false, af[mi].v, false, bfr[ni].v,
                    (short)0, acc[mi][ni], false, false);
    }

    // epilogue: C/D layout -> lane holds N=lm, VGPR r -> M = r + 8*hi
#pragma unroll
    for (int ni = 0; ni < 2; ++ni) {
        const int col = nBase + waveN * 32 + ni * 16 + lm;
        if (col >= Nstore) continue;
        const float bv = bias[col];
#pragma unroll
        for (int mi = 0; mi < 2; ++mi) {
            const int rowBase = mBase + waveM * 32 + mi * 16 + 8 * hi;
#pragma unroll
            for (int r = 0; r < 8; ++r) {
                float v = acc[mi][ni][r] + bv;
                if (RELU) v = fmaxf(v, 0.f);
                const size_t o = (size_t)(rowBase + r) * ldc + col;
                if (OUT_BF16) ((unsigned short*)Cout)[o] = f2bf(v);
                else          ((float*)Cout)[o] = v;
            }
        }
    }
}

// ---------------------------------------------------------------------------
extern "C" void kernel_launch(void* const* d_in, const int* in_sizes, int n_in,
                              void* d_out, int out_size, void* d_ws, size_t ws_size,
                              hipStream_t stream) {
    const float* features = (const float*)d_in[0];   // [16,64,64,768]
    const float* boxes    = (const float*)d_in[1];   // [16,32,4]
    const float* scores   = (const float*)d_in[2];   // [16,32]
    const float* w1       = (const float*)d_in[3];   // [1536,512]
    const float* b1       = (const float*)d_in[4];   // [512]
    const float* w2       = (const float*)d_in[5];   // [512,256]
    const float* b2       = (const float*)d_in[6];   // [256]
    const float* w3       = (const float*)d_in[7];   // [256,117]
    const float* b3       = (const float*)d_in[8];   // [117]
    // d_in[9] = labels (unused by reference output)

    // ---- workspace layout ----
    char* ws = (char*)d_ws;
    size_t off = 0;
    auto take = [&](size_t bytes) { size_t o = off; off = (off + bytes + 255) & ~(size_t)255; return o; };
    float*          pooled = (float*)         (ws + take((size_t)B_ * D_ * C_ * 4));      // 1.5 MB
    int*            hidx   = (int*)           (ws + take((size_t)B_ * NH_ * 4));
    int*            oidx   = (int*)           (ws + take((size_t)B_ * NO_ * 4));
    unsigned short* X      = (unsigned short*)(ws + take((size_t)MROWS * 1536 * 2));      // 9.4 MB
    unsigned short* h1     = (unsigned short*)(ws + take((size_t)MROWS * 512 * 2));       // 3.1 MB
    unsigned short* h2     = (unsigned short*)(ws + take((size_t)MROWS * 256 * 2));       // 1.6 MB
    unsigned short* w1t    = (unsigned short*)(ws + take((size_t)512 * 1536 * 2));        // 1.6 MB
    unsigned short* w2t    = (unsigned short*)(ws + take((size_t)256 * 512 * 2));
    unsigned short* w3t    = (unsigned short*)(ws + take((size_t)128 * 256 * 2));

    // 1) ROI mean pooling
    roi_pool_kernel<<<B_ * D_, 256, 0, stream>>>(features, boxes, pooled);

    // 2) per-group argsort
    argsort_kernel<<<B_, 32, 0, stream>>>(scores, hidx, oidx);

    // 3) weight transpose+convert (N-major bf16) and pair construction
    convert_wT_kernel<<<(512 * 1536 + 255) / 256, 256, 0, stream>>>(w1, w1t, 1536, 512, 512);
    convert_wT_kernel<<<(256 * 512  + 255) / 256, 256, 0, stream>>>(w2, w2t, 512, 256, 256);
    convert_wT_kernel<<<(128 * 256  + 255) / 256, 256, 0, stream>>>(w3, w3t, 256, 117, 128);
    build_pairs_kernel<<<MROWS, 256, 0, stream>>>(pooled, hidx, oidx, X);

    // 4) MLP via bf16 WMMA GEMMs
    gemm_bf16_kernel<true,  true ><<<dim3(MROWS / TM, 512 / TN), 256, 0, stream>>>(
        X,  w1t, b1, h1, 1536, 512, 512);
    gemm_bf16_kernel<true,  true ><<<dim3(MROWS / TM, 256 / TN), 256, 0, stream>>>(
        h1, w2t, b2, h2, 512, 256, 256);
    gemm_bf16_kernel<false, false><<<dim3(MROWS / TM, 128 / TN), 256, 0, stream>>>(
        h2, w3t, b3, (float*)d_out, 256, 117, 117);
}
